// AGCN_54692113547785
// MI455X (gfx1250) — matrix operand order repaired and checked by reading, not compiled, not run on hardware
//
#include <hip/hip_runtime.h>
#include <hip/hip_bf16.h>

typedef __attribute__((ext_vector_type(16))) _Float16 v16h;
typedef __attribute__((ext_vector_type(8)))  float    v8f;

#define NN     100000
#define EE     3200000
#define NFEAT  512
#define NHID   256
#define NCLASS 64
#define KPROP  10
#define PAD    40   // LDS row stride in halves: 80B -> 16B-aligned stores, 4B-aligned frag reads
#define NB_SCAN ((NN + 255) / 256)

#if defined(__has_builtin)
#if __has_builtin(__builtin_amdgcn_sched_barrier)
#define SCHED_BARRIER() __builtin_amdgcn_sched_barrier(0)
#endif
#endif
#ifndef SCHED_BARRIER
#define SCHED_BARRIER()
#endif

// ---------------------------------------------------------------------------
// GEMM1: h1 = relu(x @ W1 + b1), x[N,512] f32, W1[512,256] f32 -> h1[N,256] f16
// ---------------------------------------------------------------------------
__global__ __launch_bounds__(256) void agcn_gemm1(const float* __restrict__ x,
                                                  const float* __restrict__ W1,
                                                  const float* __restrict__ b1,
                                                  _Float16* __restrict__ h1) {
    __shared__ _Float16 lA[128 * PAD];
    __shared__ _Float16 lB[64 * PAD];
    const int rowBase = blockIdx.x * 128;
    const int colBase = blockIdx.y * 64;
    const int tid  = threadIdx.x;
    const int w    = tid >> 5;
    const int lane = tid & 31;
    const int hlf  = lane >> 4;
    const int l16  = lane & 15;

    v8f acc[4];
    #pragma unroll
    for (int t = 0; t < 4; ++t) acc[t] = (v8f){0.f,0.f,0.f,0.f,0.f,0.f,0.f,0.f};

    for (int kk = 0; kk < NFEAT; kk += 32) {
        __syncthreads();
        // ---- stage A + B: issue ALL global loads first, then convert/store
        float4 va[4];
        float4 vb[2];
        #pragma unroll
        for (int i = 0; i < 4; ++i) {
            int idx = tid + i * 256;          // 1024 float4 slots
            int r = idx >> 3, seg = idx & 7;  // 8 float4 per row
            int row = rowBase + r;
            int rc = row < NN ? row : NN - 1; // clamp: spill rows are never stored
            va[i] = *(const float4*)(x + (size_t)rc * NFEAT + kk + seg * 4);
        }
        #pragma unroll
        for (int i = 0; i < 2; ++i) {
            int idx = tid + i * 256;            // 512 float4 slots (32k x 16)
            int k = idx >> 4, seg = idx & 15;
            vb[i] = *(const float4*)(W1 + (size_t)(kk + k) * NHID + colBase + seg * 4);
        }
        SCHED_BARRIER();
        #pragma unroll
        for (int i = 0; i < 4; ++i) {
            int idx = tid + i * 256;
            int r = idx >> 3, seg = idx & 7;
            union { _Float16 h[4]; uint2 u; } p;
            p.h[0] = (_Float16)va[i].x; p.h[1] = (_Float16)va[i].y;
            p.h[2] = (_Float16)va[i].z; p.h[3] = (_Float16)va[i].w;
            *(uint2*)&lA[r * PAD + seg * 4] = p.u;
        }
        #pragma unroll
        for (int i = 0; i < 2; ++i) {
            int idx = tid + i * 256;
            int k = idx >> 4, seg = idx & 15;
            lB[(seg * 4 + 0) * PAD + k] = (_Float16)vb[i].x;
            lB[(seg * 4 + 1) * PAD + k] = (_Float16)vb[i].y;
            lB[(seg * 4 + 2) * PAD + k] = (_Float16)vb[i].z;
            lB[(seg * 4 + 3) * PAD + k] = (_Float16)vb[i].w;
        }
        __syncthreads();

        union { unsigned int u[8]; v16h v; } a;
        const int ar = w * 16 + l16;
        #pragma unroll
        for (int vv = 0; vv < 8; ++vv) {
            int k0 = ((vv & 3) << 1) + ((vv >> 2) << 4) + (hlf << 3);
            a.u[vv] = *(const unsigned int*)&lA[ar * PAD + k0];
        }
        #pragma unroll
        for (int t = 0; t < 4; ++t) {
            union { unsigned int u[8]; v16h v; } b;
            const int bn = t * 16 + l16;
            #pragma unroll
            for (int vv = 0; vv < 8; ++vv) {
                int k0 = ((vv & 3) << 1) + ((vv >> 2) << 4) + (hlf << 3);
                b.u[vv] = *(const unsigned int*)&lB[bn * PAD + k0];
            }
            acc[t] = __builtin_amdgcn_wmma_f32_16x16x32_f16(
                false, a.v, false, b.v, (short)0, acc[t], false, false);
        }
    }
    #pragma unroll
    for (int t = 0; t < 4; ++t) {
        int col = colBase + t * 16 + l16;
        float bias = b1[col];
        #pragma unroll
        for (int r = 0; r < 8; ++r) {
            int row = rowBase + w * 16 + r + hlf * 8;
            if (row < NN) {
                float v = acc[t][r] + bias;
                h1[(size_t)row * NHID + col] = (_Float16)fmaxf(v, 0.f);
            }
        }
    }
}

// ---------------------------------------------------------------------------
// GEMM2: inp = h1 @ W2 + b2; A tile staged with async global->LDS DMA (ASYNCcnt)
// ---------------------------------------------------------------------------
__global__ __launch_bounds__(256) void agcn_gemm2(const _Float16* __restrict__ h1,
                                                  const float* __restrict__ W2,
                                                  const float* __restrict__ b2,
                                                  float* __restrict__ inp) {
    __shared__ _Float16 lA[128 * PAD];
    __shared__ _Float16 lB[64 * PAD];
    const int rowBase = blockIdx.x * 128;
    const int tid  = threadIdx.x;
    const int w    = tid >> 5;
    const int lane = tid & 31;
    const int hlf  = lane >> 4;
    const int l16  = lane & 15;

    v8f acc[4];
    #pragma unroll
    for (int t = 0; t < 4; ++t) acc[t] = (v8f){0.f,0.f,0.f,0.f,0.f,0.f,0.f,0.f};

    for (int kk = 0; kk < NHID; kk += 32) {
        __syncthreads();
        // ---- stage A via async LDS DMA: 128 rows x 64B, b128 per lane
        #pragma unroll
        for (int i = 0; i < 2; ++i) {
            int idx = tid + i * 256;            // 512 b128 slots (128 rows x 4)
            int r = idx >> 2, seg = idx & 3;    // seg covers 8 halves (16B)
            int row = rowBase + r;
            int rc = row < NN ? row : NN - 1;   // clamp: spill rows never stored
            const _Float16* gp = h1 + (size_t)rc * NHID + kk + seg * 8;
            unsigned ldsOff = (unsigned)(size_t)&lA[r * PAD + seg * 8];
            asm volatile("global_load_async_to_lds_b128 %0, %1, off"
                         :: "v"(ldsOff), "v"((unsigned long long)(size_t)gp)
                         : "memory");
        }
        // ---- stage B transposed: W2[kk+k][n] -> lB[n][k]
        {
            float4 vb[2];
            #pragma unroll
            for (int i = 0; i < 2; ++i) {
                int idx = tid + i * 256;            // 512 float4 slots (32k x 16)
                int k = idx >> 4, seg = idx & 15;
                vb[i] = *(const float4*)(W2 + (size_t)(kk + k) * NCLASS + seg * 4);
            }
            SCHED_BARRIER();
            #pragma unroll
            for (int i = 0; i < 2; ++i) {
                int idx = tid + i * 256;
                int k = idx >> 4, seg = idx & 15;
                lB[(seg * 4 + 0) * PAD + k] = (_Float16)vb[i].x;
                lB[(seg * 4 + 1) * PAD + k] = (_Float16)vb[i].y;
                lB[(seg * 4 + 2) * PAD + k] = (_Float16)vb[i].z;
                lB[(seg * 4 + 3) * PAD + k] = (_Float16)vb[i].w;
            }
        }
        asm volatile("s_wait_asynccnt 0x0" ::: "memory");
        __syncthreads();

        union { unsigned int u[8]; v16h v; } a;
        const int ar = w * 16 + l16;
        #pragma unroll
        for (int vv = 0; vv < 8; ++vv) {
            int k0 = ((vv & 3) << 1) + ((vv >> 2) << 4) + (hlf << 3);
            a.u[vv] = *(const unsigned int*)&lA[ar * PAD + k0];
        }
        #pragma unroll
        for (int t = 0; t < 4; ++t) {
            union { unsigned int u[8]; v16h v; } b;
            const int bn = t * 16 + l16;
            #pragma unroll
            for (int vv = 0; vv < 8; ++vv) {
                int k0 = ((vv & 3) << 1) + ((vv >> 2) << 4) + (hlf << 3);
                b.u[vv] = *(const unsigned int*)&lB[bn * PAD + k0];
            }
            acc[t] = __builtin_amdgcn_wmma_f32_16x16x32_f16(
                false, a.v, false, b.v, (short)0, acc[t], false, false);
        }
    }
    #pragma unroll
    for (int t = 0; t < 4; ++t) {
        int col = t * 16 + l16;
        float bias = b2[col];
        #pragma unroll
        for (int r = 0; r < 8; ++r) {
            int row = rowBase + w * 16 + r + hlf * 8;
            if (row < NN) inp[(size_t)row * NCLASS + col] = acc[t][r] + bias;
        }
    }
}

// ---------------------------------------------------------------------------
// Prep kernels
// ---------------------------------------------------------------------------
__global__ void agcn_zero_f32(float* __restrict__ p, int n) {
    int i = blockIdx.x * blockDim.x + threadIdx.x;
    if (i < n) p[i] = 0.f;
}

__global__ void agcn_zero_i32(int* __restrict__ p, int n) {
    int i = blockIdx.x * blockDim.x + threadIdx.x;
    if (i < n) p[i] = 0;
}

// degree by SRC (matches reference segment_sum over src)
__global__ void agcn_count_deg(const long long* __restrict__ ei, float* __restrict__ deg) {
    int e = blockIdx.x * blockDim.x + threadIdx.x;
    if (e < EE) atomicAdd(&deg[(int)ei[e]], 1.0f);
}

__global__ void agcn_make_dinv(float* __restrict__ dinv) {
    int i = blockIdx.x * blockDim.x + threadIdx.x;
    if (i < NN) {
        float dg = dinv[i];
        dinv[i] = (dg > 0.f) ? rsqrtf(dg) : 0.f;
    }
}

// in-degree count by DST (for CSR)
__global__ void agcn_count_dst(const long long* __restrict__ ei, int* __restrict__ cnt) {
    int e = blockIdx.x * blockDim.x + threadIdx.x;
    if (e < EE) atomicAdd(&cnt[(int)ei[(size_t)EE + e]], 1);
}

// ---- 3-kernel exclusive scan over cnt[NN] -> rowptr[NN+1]
__global__ void agcn_scan_blocks(const int* __restrict__ cnt, int* __restrict__ bsum) {
    __shared__ int sh[256];
    int tid = threadIdx.x;
    int i = blockIdx.x * 256 + tid;
    sh[tid] = (i < NN) ? cnt[i] : 0;
    __syncthreads();
    for (int o = 128; o > 0; o >>= 1) {
        if (tid < o) sh[tid] += sh[tid + o];
        __syncthreads();
    }
    if (tid == 0) bsum[blockIdx.x] = sh[0];
}

__global__ void agcn_scan_bsum(int* __restrict__ bsum) {
    if (blockIdx.x == 0 && threadIdx.x == 0) {
        int acc = 0;
        for (int i = 0; i < NB_SCAN; ++i) { int t = bsum[i]; bsum[i] = acc; acc += t; }
    }
}

__global__ void agcn_scan_apply(const int* __restrict__ cnt, const int* __restrict__ bsum,
                                int* __restrict__ rowptr) {
    __shared__ int sh[256];
    int tid = threadIdx.x;
    int i = blockIdx.x * 256 + tid;
    int v = (i < NN) ? cnt[i] : 0;
    sh[tid] = v;
    __syncthreads();
    #pragma unroll
    for (int o = 1; o < 256; o <<= 1) {
        int t = (tid >= o) ? sh[tid - o] : 0;
        __syncthreads();
        sh[tid] += t;
        __syncthreads();
    }
    if (i < NN) rowptr[i] = sh[tid] - v + bsum[blockIdx.x];   // exclusive
    if (i == 0) rowptr[NN] = EE;
}

// fill CSR (by dst): csr_src + fused edge weight  norm = -dinv[src]*dinv[dst]
__global__ void agcn_fill_csr(const long long* __restrict__ ei,
                              const int* __restrict__ rowptr, int* __restrict__ cursor,
                              const float* __restrict__ dinv,
                              int* __restrict__ csr_src, float* __restrict__ csr_norm) {
    int e = blockIdx.x * blockDim.x + threadIdx.x;
    if (e >= EE) return;
    int s = (int)ei[e];
    int d = (int)ei[(size_t)EE + e];
    int pos = rowptr[d] + atomicAdd(&cursor[d], 1);
    csr_src[pos] = s;
    csr_norm[pos] = -dinv[s] * dinv[d];
}

// hidden0 = sigmoid(inp . scores[0] + sbias[0]) * inp   (one wave per node)
__global__ void agcn_init_hidden(const float* __restrict__ inp,
                                 float* __restrict__ hidden,
                                 const float* __restrict__ scores,
                                 const float* __restrict__ sbias) {
    int node = (blockIdx.x * blockDim.x + threadIdx.x) >> 5;
    int lane = threadIdx.x & 31;
    if (node >= NN) return;
    const float2 v = *(const float2*)(inp + (size_t)node * NCLASS + lane * 2);
    float partial = v.x * scores[lane * 2] + v.y * scores[lane * 2 + 1];
    #pragma unroll
    for (int off = 16; off > 0; off >>= 1) partial += __shfl_xor(partial, off);
    float si = 1.f / (1.f + expf(-(partial + sbias[0])));
    float2 hd; hd.x = si * v.x; hd.y = si * v.y;
    *(float2*)(hidden + (size_t)node * NCLASS + lane * 2) = hd;
}

// Fused propagation step (one wave per dst node, zero atomics):
//   scat = sum_e norm_e * inpOld[src_e]; mid = scat + inpOld[node]
//   v = inpOld[node] - t*mid; si = sigmoid(v . scores[k+1] + sbias[k+1])
//   hidden += si*v; inpNew[node] = v
__global__ void agcn_prop(const float* __restrict__ inpOld, float* __restrict__ inpNew,
                          float* __restrict__ hidden,
                          const int* __restrict__ rowptr, const int* __restrict__ csr_src,
                          const float* __restrict__ csr_norm,
                          const float* __restrict__ temp, const float* __restrict__ scores,
                          const float* __restrict__ sbias, int k) {
    int node = (blockIdx.x * blockDim.x + threadIdx.x) >> 5;
    int lane = threadIdx.x & 31;
    if (node >= NN) return;
    float t = tanhf(temp[k]);
    const float* sc = scores + (size_t)(k + 1) * NCLASS;
    float sb = sbias[k + 1];
    int beg = rowptr[node], end = rowptr[node + 1];
    float2 acc; acc.x = 0.f; acc.y = 0.f;
    for (int i = beg; i < end; ++i) {
        int s = csr_src[i];
        float nv = csr_norm[i];
        const float2 nbv = *(const float2*)(inpOld + (size_t)s * NCLASS + lane * 2);
        acc.x += nv * nbv.x;
        acc.y += nv * nbv.y;
    }
    float2 v = *(const float2*)(inpOld + (size_t)node * NCLASS + lane * 2);
    v.x = v.x - t * (acc.x + v.x);
    v.y = v.y - t * (acc.y + v.y);
    float partial = v.x * sc[lane * 2] + v.y * sc[lane * 2 + 1];
    #pragma unroll
    for (int off = 16; off > 0; off >>= 1) partial += __shfl_xor(partial, off);
    float si = 1.f / (1.f + expf(-(partial + sb)));
    float2 hd = *(float2*)(hidden + (size_t)node * NCLASS + lane * 2);
    hd.x += si * v.x;
    hd.y += si * v.y;
    *(float2*)(hidden + (size_t)node * NCLASS + lane * 2) = hd;
    *(float2*)(inpNew + (size_t)node * NCLASS + lane * 2) = v;
}

// log_softmax over 64 classes, one wave per node
__global__ void agcn_logsoftmax(const float* __restrict__ hidden, float* __restrict__ out) {
    int node = (blockIdx.x * blockDim.x + threadIdx.x) >> 5;
    int lane = threadIdx.x & 31;
    if (node >= NN) return;
    const float2 v = *(const float2*)(hidden + (size_t)node * NCLASS + lane * 2);
    float mx = fmaxf(v.x, v.y);
    #pragma unroll
    for (int off = 16; off > 0; off >>= 1) mx = fmaxf(mx, __shfl_xor(mx, off));
    float ex = expf(v.x - mx) + expf(v.y - mx);
    #pragma unroll
    for (int off = 16; off > 0; off >>= 1) ex += __shfl_xor(ex, off);
    float lse = mx + logf(ex);
    float2 o; o.x = v.x - lse; o.y = v.y - lse;
    *(float2*)(out + (size_t)node * NCLASS + lane * 2) = o;
}

// ---------------------------------------------------------------------------
extern "C" void kernel_launch(void* const* d_in, const int* in_sizes, int n_in,
                              void* d_out, int out_size, void* d_ws, size_t ws_size,
                              hipStream_t stream) {
    const float*     x      = (const float*)d_in[0];
    const long long* ei     = (const long long*)d_in[1];
    const float*     W1     = (const float*)d_in[2];
    const float*     b1     = (const float*)d_in[3];
    const float*     W2     = (const float*)d_in[4];
    const float*     b2     = (const float*)d_in[5];
    const float*     temp   = (const float*)d_in[6];
    const float*     scores = (const float*)d_in[7];
    const float*     sbias  = (const float*)d_in[8];
    float*           out    = (float*)d_out;

    char* ws = (char*)d_ws;
    size_t off = 0;
    auto alloc = [&](size_t bytes) -> char* {
        char* p = ws + off;
        off += (bytes + 255) & ~(size_t)255;
        return p;
    };
    _Float16* h1       = (_Float16*)alloc((size_t)NN * NHID * sizeof(_Float16));
    float*    inpA     = (float*)alloc((size_t)NN * NCLASS * sizeof(float));
    float*    inpB     = (float*)alloc((size_t)NN * NCLASS * sizeof(float));
    float*    hidden   = (float*)alloc((size_t)NN * NCLASS * sizeof(float));
    float*    dinv     = (float*)alloc((size_t)NN * sizeof(float));
    int*      cnt      = (int*)alloc((size_t)NN * sizeof(int));
    int*      rowptr   = (int*)alloc((size_t)(NN + 1) * sizeof(int));
    int*      bsum     = (int*)alloc((size_t)NB_SCAN * sizeof(int));
    int*      csr_src  = (int*)alloc((size_t)EE * sizeof(int));
    float*    csr_norm = (float*)alloc((size_t)EE * sizeof(float));

    const int rowTiles = (NN + 127) / 128;

    // MLP
    agcn_gemm1<<<dim3(rowTiles, NHID / 64), 256, 0, stream>>>(x, W1, b1, h1);
    agcn_gemm2<<<dim3(rowTiles, 1), 256, 0, stream>>>(h1, W2, b2, inpA);

    // Laplacian norm: deg (by src) -> dinv
    agcn_zero_f32<<<(NN + 255) / 256, 256, 0, stream>>>(dinv, NN);
    agcn_count_deg<<<(EE + 255) / 256, 256, 0, stream>>>(ei, dinv);
    agcn_make_dinv<<<(NN + 255) / 256, 256, 0, stream>>>(dinv);

    // CSR by dst: count -> exclusive scan -> fill
    agcn_zero_i32<<<(NN + 255) / 256, 256, 0, stream>>>(cnt, NN);
    agcn_count_dst<<<(EE + 255) / 256, 256, 0, stream>>>(ei, cnt);
    agcn_scan_blocks<<<NB_SCAN, 256, 0, stream>>>(cnt, bsum);
    agcn_scan_bsum<<<1, 1, 0, stream>>>(bsum);
    agcn_scan_apply<<<NB_SCAN, 256, 0, stream>>>(cnt, bsum, rowptr);
    agcn_zero_i32<<<(NN + 255) / 256, 256, 0, stream>>>(cnt, NN);   // reuse as cursor
    agcn_fill_csr<<<(EE + 255) / 256, 256, 0, stream>>>(ei, rowptr, cnt, dinv, csr_src, csr_norm);

    // hidden0
    agcn_init_hidden<<<(NN * 32 + 255) / 256, 256, 0, stream>>>(inpA, hidden, scores, sbias);

    // K fused propagation rounds (ping-pong inp buffers)
    float* cur = inpA;
    float* nxt = inpB;
    for (int k = 0; k < KPROP; ++k) {
        agcn_prop<<<(NN * 32 + 255) / 256, 256, 0, stream>>>(
            cur, nxt, hidden, rowptr, csr_src, csr_norm, temp, scores, sbias, k);
        float* t2 = cur; cur = nxt; nxt = t2;
    }

    agcn_logsoftmax<<<(NN * 32 + 255) / 256, 256, 0, stream>>>(hidden, out);
}
